// Model_11828339933500
// MI455X (gfx1250) — compile-verified, matrix operand order checked by reading, not run on hardware
//
#include <hip/hip_runtime.h>
#include <hip/hip_bf16.h>
#include <math.h>

// ---------------------------------------------------------------------------
// Types for CDNA5 WMMA (wave32): v16bf A/B fragments, v8f C/D accumulators.
// ---------------------------------------------------------------------------
typedef __bf16 bf16_t;
typedef __attribute__((ext_vector_type(16))) __bf16 v16bf;
typedef __attribute__((ext_vector_type(8)))  __bf16 v8bf;
typedef __attribute__((ext_vector_type(8)))  float  v8f;
typedef __attribute__((ext_vector_type(4)))  int    v4i;

#define DD    384     // D
#define DBB   768     // DB
#define FF    128     // F
#define BQ    1024    // B
#define NC    100000  // N candidates
#define CTXS  96      // context size
#define NEG_INF (-3.4e38f)

// gfx1250 async global->LDS DMA path (ASYNCcnt), guarded for toolchain support.
#if __has_builtin(__builtin_amdgcn_global_load_async_to_lds_b128) && \
    __has_builtin(__builtin_amdgcn_s_wait_asynccnt)
#define USE_ASYNC_LDS 1
#else
#define USE_ASYNC_LDS 0
#endif

typedef __attribute__((address_space(1))) v4i* as1_v4i_t;  // global 128-bit
typedef __attribute__((address_space(3))) v4i* as3_v4i_t;  // LDS 128-bit

// 16-byte global -> LDS async copy (per-lane addresses), tracked on ASYNCcnt.
static __device__ __forceinline__ void async_ld16(void* lds, const void* g) {
#if USE_ASYNC_LDS
  as1_v4i_t gp = (as1_v4i_t)(unsigned long long)g;
  as3_v4i_t lp = (as3_v4i_t)(unsigned int)(unsigned long long)lds;
  __builtin_amdgcn_global_load_async_to_lds_b128(gp, lp, 0, 0);
#else
  (void)lds; (void)g;
#endif
}
static __device__ __forceinline__ void async_wait0() {
#if USE_ASYNC_LDS
  __builtin_amdgcn_s_wait_asynccnt(0);
#endif
}

// D = A(16x32) * B(32x16) + C  via v_wmma_f32_16x16x32_bf16
static __device__ __forceinline__ v8f wmma_bf16(v16bf a, v16bf b, v8f c) {
  return __builtin_amdgcn_wmma_f32_16x16x32_bf16(false, a, false, b, (short)0, c,
                                                 false, false);
}

// B fragment from pre-packed weights: tile (kt, nt) stored as 512 bf16,
// lane-major (32 lanes x 16 bf16 = one contiguous 32B load per lane).
static __device__ __forceinline__ v16bf load_b_frag(const bf16_t* __restrict__ packed,
                                                    int kt, int nt, int Kt) {
  int lane = threadIdx.x & 31;
  return *(const v16bf*)(packed + (size_t)(nt * Kt + kt) * 512 + lane * 16);
}

// A fragment from LDS row-major bf16 [rows][ldk], 16 rows @ row0, K window k0..k0+31.
// 16-bit A layout: lanes 0-15 row M=l hold K {0..7,16..23}; lanes 16-31 hold
// K {8..15,24..31}. Two 16B ds loads per lane.
static __device__ __forceinline__ v16bf load_a_frag(const bf16_t* __restrict__ lds,
                                                    int row0, int ldk, int k0) {
  int lane = threadIdx.x & 31;
  const bf16_t* p = lds + (size_t)(row0 + (lane & 15)) * ldk + k0 + ((lane < 16) ? 0 : 8);
  v8bf lo = *(const v8bf*)p;
  v8bf hi = *(const v8bf*)(p + 16);
  return __builtin_shufflevector(lo, hi, 0, 1, 2, 3, 4, 5, 6, 7,
                                 8, 9, 10, 11, 12, 13, 14, 15);
}

// ---------------------------------------------------------------------------
// Pack f32 weight [K,N] row-major into bf16 B-fragment tiles (32x16 per tile),
// tiles ordered nt-major: tileIdx = nt*(K/32) + kt.
// ---------------------------------------------------------------------------
__global__ void pack_weight_kernel(const float* __restrict__ w, bf16_t* __restrict__ out,
                                   int K, int N) {
  int gid = blockIdx.x * blockDim.x + threadIdx.x;
  int ntiles = (K / 32) * (N / 16);
  if (gid >= ntiles * 32) return;
  int tile = gid >> 5, lane = gid & 31;
  int kt = tile % (K / 32), nt = tile / (K / 32);
  int n = nt * 16 + (lane & 15);
  v16bf v;
#pragma unroll
  for (int e = 0; e < 16; ++e) {
    int kk = (e < 8) ? (((lane < 16) ? 0 : 8) + e)
                     : (((lane < 16) ? 16 : 24) + (e - 8));
    v[e] = (bf16_t)w[(size_t)(kt * 32 + kk) * N + n];
  }
  *(v16bf*)(out + (size_t)tile * 512 + lane * 16) = v;
}

// ---------------------------------------------------------------------------
// 64-row tile GEMM over all N columns: C = (LN?(A)) @ Wp + bias.
// A f32 [M,K]. grid ceil(M/64), block 128 (4 waves): wave = 16-row strip,
// walking N/64 groups of 4 n-tiles (A fragment reused across 4 WMMAs).
// ---------------------------------------------------------------------------
__global__ __launch_bounds__(128) void gemm64_kernel(
    const float* __restrict__ A, const bf16_t* __restrict__ Wp,
    const float* __restrict__ bias, const float* __restrict__ ln_g,
    const float* __restrict__ ln_b, float* __restrict__ C,
    int M, int K, int N, int prenorm) {
  extern __shared__ char smem[];
  float*  xf = (float*)smem;              // [64*K] f32
  bf16_t* xa = (bf16_t*)(xf + 64 * K);    // [64*K] bf16
  float*  st = (float*)(xa + 64 * K);     // [64*2] mu, rsig
  const int tid = threadIdx.x, wave = tid >> 5, lane = tid & 31;
  const int m0 = blockIdx.x * 64;

#if USE_ASYNC_LDS
  if (m0 + 64 <= M) {
    const int CH = K / 4;  // 16B chunks per row
    for (int t = tid; t < 64 * CH; t += 128) {
      int r = t / CH, c = t - r * CH;
      async_ld16(xf + r * K + c * 4, A + (size_t)(m0 + r) * K + c * 4);
    }
    async_wait0();
  } else
#endif
  {
    for (int i = tid; i < 64 * K; i += 128) {
      int r = i / K, k = i - r * K;
      int gr = m0 + r;
      xf[i] = (gr < M) ? A[(size_t)gr * K + k] : 0.0f;
    }
  }
  __syncthreads();
  if (prenorm) {
    for (int r = tid; r < 64; r += 128) {
      float s = 0.f;
      for (int k = 0; k < K; ++k) s += xf[r * K + k];
      float mu = s / (float)K;
      float v = 0.f;
      for (int k = 0; k < K; ++k) { float c = xf[r * K + k] - mu; v += c * c; }
      st[2 * r] = mu;
      st[2 * r + 1] = rsqrtf(v / (float)K + 1e-5f);
    }
    __syncthreads();
  }
  for (int i = tid; i < 64 * K; i += 128) {
    int r = i / K, k = i - r * K;
    float v = xf[i];
    if (prenorm) v = (v - st[2 * r]) * st[2 * r + 1] * ln_g[k] + ln_b[k];
    xa[i] = (bf16_t)v;
  }
  __syncthreads();

  const int Kt = K >> 5;
  for (int g4 = 0; g4 < N / 64; ++g4) {
    v8f acc[4];
#pragma unroll
    for (int nt = 0; nt < 4; ++nt)
#pragma unroll
      for (int i = 0; i < 8; ++i) acc[nt][i] = 0.f;
    for (int kt = 0; kt < Kt; ++kt) {
      v16bf af = load_a_frag(xa, wave * 16, K, kt * 32);
#pragma unroll
      for (int nt = 0; nt < 4; ++nt) {
        v16bf bf = load_b_frag(Wp, kt, g4 * 4 + nt, Kt);
        acc[nt] = wmma_bf16(af, bf, acc[nt]);
      }
    }
#pragma unroll
    for (int nt = 0; nt < 4; ++nt)
#pragma unroll
      for (int i = 0; i < 8; ++i) {
        int row = wave * 16 + i + ((lane < 16) ? 0 : 8);
        int gr = m0 + row;
        int col = g4 * 64 + nt * 16 + (lane & 15);
        if (gr < M) C[(size_t)gr * N + col] = acc[nt][i] + bias[col];
      }
  }
}

// ---------------------------------------------------------------------------
// Fused residual MLP block: Y = X + relu(LN?(X) @ W1 + b1) @ W2 + b2.
// 64 rows / block, 256 threads (8 waves). Whole hidden (64x768 bf16) in LDS.
// ---------------------------------------------------------------------------
__global__ __launch_bounds__(256) void fused_mlp_kernel(
    const float* __restrict__ X, float* __restrict__ Y,
    const bf16_t* __restrict__ W1p, const float* __restrict__ b1,
    const bf16_t* __restrict__ W2p, const float* __restrict__ b2,
    const float* __restrict__ ln_g, const float* __restrict__ ln_b,
    int M, int prenorm) {
  extern __shared__ char smem[];
  float*  xres = (float*)smem;                 // [64*384] f32 (residual)
  bf16_t* xa   = (bf16_t*)(xres + 64 * DD);    // [64*384] bf16
  bf16_t* h    = xa + 64 * DD;                 // [64*768] bf16 hidden
  float*  st   = (float*)(h + 64 * DBB);       // [64*2]
  const int tid = threadIdx.x, wave = tid >> 5, lane = tid & 31;
  const int m0 = blockIdx.x * 64;

#if USE_ASYNC_LDS
  if (m0 + 64 <= M) {
    const int CH = DD / 4;
    for (int t = tid; t < 64 * CH; t += 256) {
      int r = t / CH, c = t - r * CH;
      async_ld16(xres + r * DD + c * 4, X + (size_t)(m0 + r) * DD + c * 4);
    }
    async_wait0();
  } else
#endif
  {
    for (int i = tid; i < 64 * DD; i += 256) {
      int r = i / DD, k = i - r * DD;
      int gr = m0 + r;
      xres[i] = (gr < M) ? X[(size_t)gr * DD + k] : 0.0f;
    }
  }
  __syncthreads();
  if (prenorm) {
    for (int r = tid; r < 64; r += 256) {
      float s = 0.f;
      for (int k = 0; k < DD; ++k) s += xres[r * DD + k];
      float mu = s / (float)DD;
      float v = 0.f;
      for (int k = 0; k < DD; ++k) { float c = xres[r * DD + k] - mu; v += c * c; }
      st[2 * r] = mu;
      st[2 * r + 1] = rsqrtf(v / (float)DD + 1e-5f);
    }
    __syncthreads();
  }
  for (int i = tid; i < 64 * DD; i += 256) {
    int r = i / DD, k = i - r * DD;
    float v = xres[i];
    if (prenorm) v = (v - st[2 * r]) * st[2 * r + 1] * ln_g[k] + ln_b[k];
    xa[i] = (bf16_t)v;
  }
  __syncthreads();

  // Stage 1: h = relu(xa @ W1 + b1), output 64x768, wave w owns 96 columns.
  for (int nt = 0; nt < 6; ++nt) {
    int ntg = wave * 6 + nt;
    v8f acc[4];
#pragma unroll
    for (int m = 0; m < 4; ++m)
#pragma unroll
      for (int i = 0; i < 8; ++i) acc[m][i] = 0.f;
    for (int kt = 0; kt < 12; ++kt) {
      v16bf bf = load_b_frag(W1p, kt, ntg, 12);
#pragma unroll
      for (int m = 0; m < 4; ++m) {
        v16bf af = load_a_frag(xa, m * 16, DD, kt * 32);
        acc[m] = wmma_bf16(af, bf, acc[m]);
      }
    }
#pragma unroll
    for (int m = 0; m < 4; ++m)
#pragma unroll
      for (int i = 0; i < 8; ++i) {
        int row = m * 16 + i + ((lane < 16) ? 0 : 8);
        int col = ntg * 16 + (lane & 15);
        float v = acc[m][i] + b1[col];
        h[row * DBB + col] = (bf16_t)fmaxf(v, 0.f);
      }
  }
  __syncthreads();

  // Stage 2: Y = h @ W2 + b2 + residual, output 64x384, wave w owns 48 cols.
  for (int nt = 0; nt < 3; ++nt) {
    int ntg = wave * 3 + nt;
    v8f acc[4];
#pragma unroll
    for (int m = 0; m < 4; ++m)
#pragma unroll
      for (int i = 0; i < 8; ++i) acc[m][i] = 0.f;
    for (int kt = 0; kt < 24; ++kt) {
      v16bf bf = load_b_frag(W2p, kt, ntg, 24);
#pragma unroll
      for (int m = 0; m < 4; ++m) {
        v16bf af = load_a_frag(h, m * 16, DBB, kt * 32);
        acc[m] = wmma_bf16(af, bf, acc[m]);
      }
    }
#pragma unroll
    for (int m = 0; m < 4; ++m)
#pragma unroll
      for (int i = 0; i < 8; ++i) {
        int row = m * 16 + i + ((lane < 16) ? 0 : 8);
        int gr = m0 + row;
        int col = ntg * 16 + (lane & 15);
        if (gr < M) Y[(size_t)gr * DD + col] = acc[m][i] + b2[col] + xres[row * DD + col];
      }
  }
}

// ---------------------------------------------------------------------------
// Per-row squared norm of cand_k.
// ---------------------------------------------------------------------------
__global__ void rownorm2_kernel(const float* __restrict__ ck, float* __restrict__ n2,
                                int M) {
  int r = blockIdx.x * blockDim.x + threadIdx.x;
  if (r >= M) return;
  const float4* p = (const float4*)(ck + (size_t)r * DD);
  float s = 0.f;
#pragma unroll 4
  for (int i = 0; i < DD / 4; ++i) {
    float4 v = p[i];
    s += v.x * v.x + v.y * v.y + v.z * v.z + v.w * v.w;
  }
  n2[r] = s;
}

// ---------------------------------------------------------------------------
// Brute-force L2 kNN (cand_k is L2-cache-resident: 153.6MB < 192MB).
// One block per query row (128 threads). Each thread keeps a private top-96,
// merged in LDS, then 96 sorted argmax extractions (== top_k order) + softmax.
// ---------------------------------------------------------------------------
__global__ __launch_bounds__(128) void knn_topk_kernel(
    const float* __restrict__ qk, const float* __restrict__ ck,
    const float* __restrict__ n2, const int* __restrict__ cand_idx,
    int N, int* __restrict__ ctx_idx, float* __restrict__ probs_ws,
    int* __restrict__ absidx_out, float* __restrict__ probs_out) {
  extern __shared__ char smem[];
  float* kq   = (float*)smem;                 // 384
  float* mv   = kq + DD;                      // 128*96
  int*   mi   = (int*)(mv + 128 * CTXS);      // 128*96
  float* redv = (float*)(mi + 128 * CTXS);    // 128
  int*   redi = (int*)(redv + 128);           // 128
  float* selv = (float*)(redi + 128);         // 96
  int*   seli = (int*)(selv + CTXS);          // 96
  float* pe   = (float*)(seli + CTXS);        // 97
  const int b = blockIdx.x, tid = threadIdx.x;

  for (int d = tid; d < DD; d += 128) kq[d] = qk[(size_t)b * DD + d];
  __syncthreads();

  float bd[CTXS];
  int   bi[CTXS];
  int cnt = 0;
  float minv = NEG_INF;
  int minp = 0;
  const float4* kq4 = (const float4*)kq;
  for (int j = tid; j < N; j += 128) {
    if (j + 128 < N) __builtin_prefetch(ck + (size_t)(j + 128) * DD, 0, 0);
    const float4* cr = (const float4*)(ck + (size_t)j * DD);
    float s = 0.f;
#pragma unroll 8
    for (int i = 0; i < DD / 4; ++i) {
      float4 a = kq4[i], c = cr[i];
      s += a.x * c.x + a.y * c.y + a.z * c.z + a.w * c.w;
    }
    float sim = 2.f * s - n2[j];  // == sims + ||k||^2 (constant per row)
    if (cnt < CTXS) {
      bd[cnt] = sim; bi[cnt] = j; ++cnt;
      if (cnt == CTXS) {
        minv = bd[0]; minp = 0;
        for (int i = 1; i < CTXS; ++i) if (bd[i] < minv) { minv = bd[i]; minp = i; }
      }
    } else if (sim > minv) {
      bd[minp] = sim; bi[minp] = j;
      minv = bd[0]; minp = 0;
      for (int i = 1; i < CTXS; ++i) if (bd[i] < minv) { minv = bd[i]; minp = i; }
    }
  }
  for (int i = 0; i < CTXS; ++i) {
    mv[tid * CTXS + i] = (i < cnt) ? bd[i] : NEG_INF;
    mi[tid * CTXS + i] = (i < cnt) ? bi[i] : 0;
  }
  __syncthreads();

  for (int s = 0; s < CTXS; ++s) {
    float lm = NEG_INF;
    int lp = tid * CTXS;
    for (int i = 0; i < CTXS; ++i) {
      float v = mv[tid * CTXS + i];
      if (v > lm) { lm = v; lp = tid * CTXS + i; }
    }
    redv[tid] = lm; redi[tid] = lp;
    __syncthreads();
    for (int off = 64; off > 0; off >>= 1) {
      if (tid < off && redv[tid + off] > redv[tid]) {
        redv[tid] = redv[tid + off];
        redi[tid] = redi[tid + off];
      }
      __syncthreads();
    }
    if (tid == 0) {
      int pos = redi[0];
      selv[s] = redv[0];
      seli[s] = mi[pos];
      mv[pos] = NEG_INF;
    }
    __syncthreads();
  }
  // softmax (shift-invariant; selv[0] is the max since sorted descending)
  if (tid < CTXS) pe[tid] = expf(selv[tid] - selv[0]);
  __syncthreads();
  if (tid == 0) {
    float s = 0.f;
    for (int i = 0; i < CTXS; ++i) s += pe[i];
    pe[CTXS] = s;
  }
  __syncthreads();
  if (tid < CTXS) {
    float p = pe[tid] / pe[CTXS];
    int ci = seli[tid];
    ctx_idx[b * CTXS + tid] = ci;
    probs_ws[b * CTXS + tid] = p;
    absidx_out[b * CTXS + tid] = cand_idx[ci];
    probs_out[b * CTXS + tid] = p;
  }
}

// ---------------------------------------------------------------------------
// Context values + weighted aggregation, one block (256 thr) per query:
// X[b] += sum_s probs[s] * ( y_s*Wl + bl + relu((k-ck_s)@Wt1+bt1)@Wt2 )
// The gathered cand_k rows are async-DMA'd into LDS; that staging region is
// reused for the bf16 hidden matrix.
// ---------------------------------------------------------------------------
__global__ __launch_bounds__(256) void context_kernel(
    const float* __restrict__ qk, const float* __restrict__ ck,
    const float* __restrict__ cand_y, const int* __restrict__ ctx_idx,
    const float* __restrict__ probs, const float* __restrict__ Wl,
    const float* __restrict__ bl, const bf16_t* __restrict__ Wt1p,
    const float* __restrict__ bt1, const bf16_t* __restrict__ Wt2p,
    float* __restrict__ X) {
  extern __shared__ char smem[];
  bf16_t* h    = (bf16_t*)smem;               // [96*768] bf16 (hidden)
  float*  ckst = (float*)smem;                // alias: [96*384] f32 staging
  bf16_t* a    = (bf16_t*)(smem + (size_t)CTXS * DBB * 2);  // [96*384] bf16
  float*  kq   = (float*)(a + CTXS * DD);     // 384
  float*  ctxx = kq + DD;                     // 384
  float*  pr   = ctxx + DD;                   // 96
  float*  yy   = pr + CTXS;                   // 96
  int*    idx  = (int*)(yy + CTXS);           // 96
  const int b = blockIdx.x, tid = threadIdx.x, wave = tid >> 5, lane = tid & 31;

  if (tid < CTXS) {
    int ci = ctx_idx[b * CTXS + tid];
    idx[tid] = ci;
    pr[tid] = probs[b * CTXS + tid];
    yy[tid] = cand_y[ci];
  }
  for (int d = tid; d < DD; d += 256) {
    kq[d] = qk[(size_t)b * DD + d];
    ctxx[d] = 0.f;
  }
  __syncthreads();
#if USE_ASYNC_LDS
  {
    const int CH = DD / 4;
    for (int t = tid; t < CTXS * CH; t += 256) {
      int s = t / CH, c = t - s * CH;
      async_ld16(ckst + s * DD + c * 4, ck + (size_t)idx[s] * DD + c * 4);
    }
    async_wait0();
    __syncthreads();
    for (int t = tid; t < CTXS * DD; t += 256) {
      int s = t / DD, d = t - s * DD;
      a[t] = (bf16_t)(kq[d] - ckst[s * DD + d]);
    }
  }
#else
  for (int t = tid; t < CTXS * DD; t += 256) {
    int s = t / DD, d = t - s * DD;
    a[t] = (bf16_t)(kq[d] - ck[(size_t)idx[s] * DD + d]);
  }
#endif
  __syncthreads();

  // Stage 1: h = relu(a @ Wt1 + bt1)  [96 x 768]  (overwrites staging region)
  for (int nt = 0; nt < 6; ++nt) {
    int ntg = wave * 6 + nt;
    v8f acc[6];
#pragma unroll
    for (int m = 0; m < 6; ++m)
#pragma unroll
      for (int i = 0; i < 8; ++i) acc[m][i] = 0.f;
    for (int kt = 0; kt < 12; ++kt) {
      v16bf bf = load_b_frag(Wt1p, kt, ntg, 12);
#pragma unroll
      for (int m = 0; m < 6; ++m) {
        v16bf af = load_a_frag(a, m * 16, DD, kt * 32);
        acc[m] = wmma_bf16(af, bf, acc[m]);
      }
    }
#pragma unroll
    for (int m = 0; m < 6; ++m)
#pragma unroll
      for (int i = 0; i < 8; ++i) {
        int row = m * 16 + i + ((lane < 16) ? 0 : 8);
        int col = ntg * 16 + (lane & 15);
        float v = acc[m][i] + bt1[col];
        h[(size_t)row * DBB + col] = (bf16_t)fmaxf(v, 0.f);
      }
  }
  __syncthreads();

  // Stage 2: v = h @ Wt2; ctxx[col] += sum_s pr[s]*v[s,col] via ds_add_f32
  for (int nt = 0; nt < 3; ++nt) {
    int ntg = wave * 3 + nt;
    v8f acc[6];
#pragma unroll
    for (int m = 0; m < 6; ++m)
#pragma unroll
      for (int i = 0; i < 8; ++i) acc[m][i] = 0.f;
    for (int kt = 0; kt < 24; ++kt) {
      v16bf bf = load_b_frag(Wt2p, kt, ntg, 24);
#pragma unroll
      for (int m = 0; m < 6; ++m) {
        v16bf af = load_a_frag(h, m * 16, DBB, kt * 32);
        acc[m] = wmma_bf16(af, bf, acc[m]);
      }
    }
#pragma unroll
    for (int m = 0; m < 6; ++m) {
      float part = 0.f;
#pragma unroll
      for (int i = 0; i < 8; ++i) {
        int row = m * 16 + i + ((lane < 16) ? 0 : 8);
        part += pr[row] * acc[m][i];
      }
      atomicAdd(&ctxx[ntg * 16 + (lane & 15)], part);
    }
  }
  __syncthreads();

  float py = 0.f;
  for (int s = 0; s < CTXS; ++s) py += pr[s] * yy[s];  // sum(probs)==1
  for (int d = tid; d < DD; d += 256)
    X[(size_t)b * DD + d] += ctxx[d] + py * Wl[d] + bl[d];
}

// ---------------------------------------------------------------------------
// Head: y = relu(LN(x)) @ Wh + bh, one block per row.
// ---------------------------------------------------------------------------
__global__ __launch_bounds__(128) void head_kernel(
    const float* __restrict__ X, const float* __restrict__ g,
    const float* __restrict__ bb, const float* __restrict__ Wh,
    const float* __restrict__ bh, float* __restrict__ y) {
  __shared__ float red[128];
  const int row = blockIdx.x, tid = threadIdx.x;
  const float* x = X + (size_t)row * DD;
  float s = 0.f;
  for (int d = tid; d < DD; d += 128) s += x[d];
  red[tid] = s; __syncthreads();
  for (int off = 64; off > 0; off >>= 1) { if (tid < off) red[tid] += red[tid + off]; __syncthreads(); }
  float mu = red[0] / (float)DD; __syncthreads();
  s = 0.f;
  for (int d = tid; d < DD; d += 128) { float c = x[d] - mu; s += c * c; }
  red[tid] = s; __syncthreads();
  for (int off = 64; off > 0; off >>= 1) { if (tid < off) red[tid] += red[tid + off]; __syncthreads(); }
  float rs = rsqrtf(red[0] / (float)DD + 1e-5f); __syncthreads();
  s = 0.f;
  for (int d = tid; d < DD; d += 128) {
    float v = (x[d] - mu) * rs * g[d] + bb[d];
    s += fmaxf(v, 0.f) * Wh[d];
  }
  red[tid] = s; __syncthreads();
  for (int off = 64; off > 0; off >>= 1) { if (tid < off) red[tid] += red[tid + off]; __syncthreads(); }
  if (tid == 0) y[row] = red[0] + bh[0];
}

__global__ void zero_kernel(float* __restrict__ p, int n) {
  int i = blockIdx.x * blockDim.x + threadIdx.x;
  if (i < n) p[i] = 0.f;
}

// ---------------------------------------------------------------------------
// Host launch
// ---------------------------------------------------------------------------
extern "C" void kernel_launch(void* const* d_in, const int* in_sizes, int n_in,
                              void* d_out, int out_size, void* d_ws, size_t ws_size,
                              hipStream_t stream) {
  (void)in_sizes; (void)n_in; (void)out_size; (void)ws_size;
  const float* x_num    = (const float*)d_in[0];
  const float* cand_x   = (const float*)d_in[1];
  const float* cand_y   = (const float*)d_in[2];
  const int*   cand_idx = (const int*)d_in[3];
  const float* W_in  = (const float*)d_in[5];
  const float* b_in  = (const float*)d_in[6];
  const float* b0_W1 = (const float*)d_in[7];
  const float* b0_b1 = (const float*)d_in[8];
  const float* b0_W2 = (const float*)d_in[9];
  const float* b0_b2 = (const float*)d_in[10];
  const float* b1_g  = (const float*)d_in[11];
  const float* b1_b  = (const float*)d_in[12];
  const float* b1_W1 = (const float*)d_in[13];
  const float* b1_b1 = (const float*)d_in[14];
  const float* b1_W2 = (const float*)d_in[15];
  const float* b1_b2 = (const float*)d_in[16];
  const float* mln_g = (const float*)d_in[17];
  const float* mln_b = (const float*)d_in[18];
  const float* Wk    = (const float*)d_in[19];
  const float* bk    = (const float*)d_in[20];
  const float* Wl    = (const float*)d_in[21];
  const float* bl    = (const float*)d_in[22];
  const float* Wt1   = (const float*)d_in[23];
  const float* bt1   = (const float*)d_in[24];
  const float* Wt2   = (const float*)d_in[25];
  const float* p0_g  = (const float*)d_in[26];
  const float* p0_b  = (const float*)d_in[27];
  const float* p0_W1 = (const float*)d_in[28];
  const float* p0_b1 = (const float*)d_in[29];
  const float* p0_W2 = (const float*)d_in[30];
  const float* p0_b2 = (const float*)d_in[31];
  const float* p1_g  = (const float*)d_in[32];
  const float* p1_b  = (const float*)d_in[33];
  const float* p1_W1 = (const float*)d_in[34];
  const float* p1_b1 = (const float*)d_in[35];
  const float* p1_W2 = (const float*)d_in[36];
  const float* p1_b2 = (const float*)d_in[37];
  const float* h_g   = (const float*)d_in[38];
  const float* h_b   = (const float*)d_in[39];
  const float* Wh    = (const float*)d_in[40];
  const float* bh    = (const float*)d_in[41];

  float* out_y    = (float*)d_out;                         // [1024]
  int*   out_idx  = (int*)(out_y + BQ);                    // [1024*96] int32 bits
  float* out_prob = (float*)(out_idx + BQ * CTXS);         // [1024*96]
  float* out_miss = out_prob + BQ * CTXS;                  // [1024]

  // Workspace carve
  char* wsb = (char*)d_ws;
  size_t off = 0;
  auto carve = [&](size_t bytes) -> char* {
    char* p = wsb + off;
    off += (bytes + 255) & ~(size_t)255;
    return p;
  };
  bf16_t* pw_Win  = (bf16_t*)carve((size_t)FF * DD * 2);
  bf16_t* pw_b0W1 = (bf16_t*)carve((size_t)DD * DBB * 2);
  bf16_t* pw_b0W2 = (bf16_t*)carve((size_t)DBB * DD * 2);
  bf16_t* pw_b1W1 = (bf16_t*)carve((size_t)DD * DBB * 2);
  bf16_t* pw_b1W2 = (bf16_t*)carve((size_t)DBB * DD * 2);
  bf16_t* pw_Wk   = (bf16_t*)carve((size_t)DD * DD * 2);
  bf16_t* pw_Wt1  = (bf16_t*)carve((size_t)DD * DBB * 2);
  bf16_t* pw_Wt2  = (bf16_t*)carve((size_t)DBB * DD * 2);
  bf16_t* pw_p0W1 = (bf16_t*)carve((size_t)DD * DBB * 2);
  bf16_t* pw_p0W2 = (bf16_t*)carve((size_t)DBB * DD * 2);
  bf16_t* pw_p1W1 = (bf16_t*)carve((size_t)DD * DBB * 2);
  bf16_t* pw_p1W2 = (bf16_t*)carve((size_t)DBB * DD * 2);
  const int MB = (NC + 63) / 64;        // 1563 row-tiles
  const int Mpad = MB * 64;             // 100032
  float* xc  = (float*)carve((size_t)Mpad * DD * 4);   // candidate activations
  float* ckb = (float*)carve((size_t)Mpad * DD * 4);   // cand_k
  float* n2  = (float*)carve((size_t)Mpad * 4);
  float* xq  = (float*)carve((size_t)BQ * DD * 4);     // query activations
  float* qkb = (float*)carve((size_t)BQ * DD * 4);     // query k
  int*   cidx = (int*)carve((size_t)BQ * CTXS * 4);
  float* pws  = (float*)carve((size_t)BQ * CTXS * 4);

  auto pack = [&](const float* w, bf16_t* dst, int K, int N) {
    int threads = (K / 32) * (N / 16) * 32;
    pack_weight_kernel<<<(threads + 255) / 256, 256, 0, stream>>>(w, dst, K, N);
  };
  pack(W_in, pw_Win, FF, DD);
  pack(b0_W1, pw_b0W1, DD, DBB);  pack(b0_W2, pw_b0W2, DBB, DD);
  pack(b1_W1, pw_b1W1, DD, DBB);  pack(b1_W2, pw_b1W2, DBB, DD);
  pack(Wk, pw_Wk, DD, DD);
  pack(Wt1, pw_Wt1, DD, DBB);     pack(Wt2, pw_Wt2, DBB, DD);
  pack(p0_W1, pw_p0W1, DD, DBB);  pack(p0_W2, pw_p0W2, DBB, DD);
  pack(p1_W1, pw_p1W1, DD, DBB);  pack(p1_W2, pw_p1W2, DBB, DD);

  const size_t sm_g128 = (size_t)64 * FF * 6 + 512;
  const size_t sm_g384 = (size_t)64 * DD * 6 + 512;
  const size_t sm_f = (size_t)64 * DD * 4 + (size_t)64 * DD * 2 + (size_t)64 * DBB * 2 + 512;

  // ---- candidate encoder ----
  gemm64_kernel<<<MB, 128, sm_g128, stream>>>(cand_x, pw_Win, b_in, nullptr, nullptr,
                                              xc, NC, FF, DD, 0);
  fused_mlp_kernel<<<MB, 256, sm_f, stream>>>(xc, xc, pw_b0W1, b0_b1, pw_b0W2, b0_b2,
                                              nullptr, nullptr, NC, 0);
  fused_mlp_kernel<<<MB, 256, sm_f, stream>>>(xc, xc, pw_b1W1, b1_b1, pw_b1W2, b1_b2,
                                              b1_g, b1_b, NC, 1);
  gemm64_kernel<<<MB, 128, sm_g384, stream>>>(xc, pw_Wk, bk, mln_g, mln_b,
                                              ckb, NC, DD, DD, 1);
  // ---- query encoder ----
  gemm64_kernel<<<BQ / 64, 128, sm_g128, stream>>>(x_num, pw_Win, b_in, nullptr, nullptr,
                                                   xq, BQ, FF, DD, 0);
  fused_mlp_kernel<<<BQ / 64, 256, sm_f, stream>>>(xq, xq, pw_b0W1, b0_b1, pw_b0W2, b0_b2,
                                                   nullptr, nullptr, BQ, 0);
  fused_mlp_kernel<<<BQ / 64, 256, sm_f, stream>>>(xq, xq, pw_b1W1, b1_b1, pw_b1W2, b1_b2,
                                                   b1_g, b1_b, BQ, 1);
  gemm64_kernel<<<BQ / 64, 128, sm_g384, stream>>>(xq, pw_Wk, bk, mln_g, mln_b,
                                                   qkb, BQ, DD, DD, 1);
  // ---- kNN ----
  rownorm2_kernel<<<(NC + 255) / 256, 256, 0, stream>>>(ckb, n2, NC);
  const size_t sm_knn = (size_t)DD * 4 + (size_t)128 * CTXS * 8 + 128 * 8 + CTXS * 8 + 512;
  knn_topk_kernel<<<BQ, 128, sm_knn, stream>>>(qkb, ckb, n2, cand_idx, NC,
                                               cidx, pws, out_idx, out_prob);
  // ---- context aggregation ----
  const size_t sm_ctx = (size_t)CTXS * DBB * 2 + (size_t)CTXS * DD * 2 +
                        (size_t)(2 * DD + 2 * CTXS) * 4 + CTXS * 4 + 256;
  context_kernel<<<BQ, 256, sm_ctx, stream>>>(qkb, ckb, cand_y, cidx, pws,
                                              Wl, bl, pw_Wt1, bt1, pw_Wt2, xq);
  // ---- predictor + head ----
  fused_mlp_kernel<<<BQ / 64, 256, sm_f, stream>>>(xq, xq, pw_p0W1, p0_b1, pw_p0W2, p0_b2,
                                                   p0_g, p0_b, BQ, 1);
  fused_mlp_kernel<<<BQ / 64, 256, sm_f, stream>>>(xq, xq, pw_p1W1, p1_b1, pw_p1W2, p1_b2,
                                                   p1_g, p1_b, BQ, 1);
  head_kernel<<<BQ, 128, 0, stream>>>(xq, h_g, h_b, Wh, bh, out_y);
  zero_kernel<<<(BQ + 255) / 256, 256, 0, stream>>>(out_miss, BQ);
}
